// MoE_67851893342555
// MI455X (gfx1250) — matrix operand order repaired and checked by reading, not compiled, not run on hardware
//
#include <hip/hip_runtime.h>
#include <hip/hip_bf16.h>
#include <math.h>

typedef __attribute__((ext_vector_type(16))) __bf16 v16bf;
typedef __attribute__((ext_vector_type(8)))  float  v8f;
typedef __attribute__((ext_vector_type(4)))  unsigned int u32x4;
typedef __attribute__((ext_vector_type(8)))  unsigned int u32x8;

#define DMODEL 512
#define DFFN   2048
#define NEXP   8
#define TOPK   2
#define KBLK   64
#define LDSROW 144                      // 64 bf16 (128B) + 16B TDM pad: conflict-free, 16B aligned
#define BUFSZ  ((128 + 64) * LDSROW)    // one double-buffer stage: A tile + B tile

static __device__ __forceinline__ unsigned short f2bf(float f) {
  unsigned u = __float_as_uint(f);
  unsigned r = u + 0x7FFFu + ((u >> 16) & 1u);   // round-to-nearest-even
  return (unsigned short)(r >> 16);
}

// ---- Tensor Data Mover: async 2D tile load (global -> LDS) via D# descriptor ----
// Tile: tileRows rows of tileK contiguous bf16 elements; row stride = kElems elements.
// LDS padding: 4 dwords (16B) after every 32 dwords (one 128B row) -> LDSROW stride.
static __device__ __forceinline__ void tdm_load_2d(unsigned ldsOff,
                                                   const unsigned short* gptr,
                                                   unsigned kElems, unsigned rows,
                                                   unsigned tileK, unsigned tileRows) {
  unsigned long long ga = (unsigned long long)(size_t)gptr;
  u32x4 g0;
  g0[0] = 1u;                                            // count=1, user desc
  g0[1] = ldsOff;                                        // lds_addr (bytes)
  g0[2] = (unsigned)ga;                                  // global_addr[31:0]
  g0[3] = ((unsigned)(ga >> 32) & 0x01FFFFFFu) | (2u << 30);  // addr[56:32], type=2
  u32x8 g1;
  g1[0] = (1u << 16)        // data_size = 1 (2 bytes)
        | (1u << 20)        // pad_enable
        | (4u << 22)        // pad_interval: 32 dwords (=128B row)
        | (3u << 25);       // pad_amount: 4 dwords (=16B)
  g1[1] = (kElems & 0xFFFFu) << 16;                      // tensor_dim0[15:0]
  g1[2] = (kElems >> 16) | ((rows & 0xFFFFu) << 16);     // tensor_dim0[31:16], dim1[15:0]
  g1[3] = (rows >> 16) | (tileK << 16);                  // dim1[31:16], tile_dim0
  g1[4] = tileRows & 0xFFFFu;                            // tile_dim1 (tile_dim2=0)
  g1[5] = kElems;                                        // tensor_dim0_stride[31:0]
  g1[6] = 0u;                                            // stride0[47:32]=0, stride1 lo=0
  g1[7] = 0u;
  asm volatile("tensor_load_to_lds %0, %1" :: "s"(g0), "s"(g1) : "memory");
}

// ---------------- Router: one wave per token ----------------
__global__ void router_kernel(const float* __restrict__ x,
                              const float* __restrict__ Wr,
                              const float* __restrict__ br,
                              int* __restrict__ tokExpert,
                              float* __restrict__ tokW,
                              int N) {
  int wave = threadIdx.x >> 5;
  int lane = threadIdx.x & 31;
  int tok  = blockIdx.x * 8 + wave;
  if (tok >= N) return;
  const float* xr = x + (size_t)tok * DMODEL;
  float acc[NEXP];
#pragma unroll
  for (int e = 0; e < NEXP; ++e) acc[e] = 0.f;
  for (int i = lane; i < DMODEL; i += 32) {
    float xv = xr[i];
    const float* wr = Wr + (size_t)i * NEXP;
#pragma unroll
    for (int e = 0; e < NEXP; ++e) acc[e] += xv * wr[e];
  }
#pragma unroll
  for (int e = 0; e < NEXP; ++e)
    for (int off = 16; off > 0; off >>= 1)
      acc[e] += __shfl_xor(acc[e], off, 32);
  if (lane == 0) {
    float p[NEXP];
    float mx = -INFINITY;
#pragma unroll
    for (int e = 0; e < NEXP; ++e) { p[e] = acc[e] + br[e]; mx = fmaxf(mx, p[e]); }
    float s = 0.f;
#pragma unroll
    for (int e = 0; e < NEXP; ++e) { p[e] = __expf(p[e] - mx); s += p[e]; }
    float inv = 1.f / s;
#pragma unroll
    for (int e = 0; e < NEXP; ++e) p[e] *= inv;
    int e0 = 0;
#pragma unroll
    for (int e = 1; e < NEXP; ++e) if (p[e] > p[e0]) e0 = e;
    int e1 = (e0 == 0) ? 1 : 0;
#pragma unroll
    for (int e = 0; e < NEXP; ++e) { if (e == e0) continue; if (p[e] > p[e1]) e1 = e; }
    float den = p[e0] + p[e1] + 1e-9f;
    tokExpert[2 * tok]     = e0;
    tokExpert[2 * tok + 1] = e1;
    tokW[2 * tok]          = p[e0] / den;
    tokW[2 * tok + 1]      = p[e1] / den;
  }
}

// ---------------- Dispatch scan (order-exact capacity) ----------------
__global__ void count_kernel(const int* __restrict__ tokExpert,
                             int* __restrict__ chunkCounts,
                             int slotsPerChunk, int nSlots) {
  int lane = threadIdx.x, chunk = blockIdx.x;
  int base = chunk * slotsPerChunk;
  int cnt[NEXP];
#pragma unroll
  for (int e = 0; e < NEXP; ++e) cnt[e] = 0;
  for (int i = base + lane; i < base + slotsPerChunk; i += 32) {
    int e = (i < nSlots) ? tokExpert[i] : -1;
#pragma unroll
    for (int ee = 0; ee < NEXP; ++ee)
      cnt[ee] += __popcll(__ballot(e == ee));
  }
  if (lane == 0)
#pragma unroll
    for (int ee = 0; ee < NEXP; ++ee) chunkCounts[chunk * NEXP + ee] = cnt[ee];
}

__global__ void scan_kernel(const int* __restrict__ chunkCounts,
                            int* __restrict__ chunkOffsets, int nChunks) {
  int e = threadIdx.x;
  if (e >= NEXP) return;
  int run = 0;
  for (int c = 0; c < nChunks; ++c) {
    chunkOffsets[c * NEXP + e] = run;
    run += chunkCounts[c * NEXP + e];
  }
}

__global__ void assign_kernel(const int* __restrict__ tokExpert,
                              const int* __restrict__ chunkOffsets,
                              int* __restrict__ slotRow,
                              int slotsPerChunk, int nSlots, int cap, int capR) {
  int lane = threadIdx.x, chunk = blockIdx.x;
  int cnt[NEXP];
#pragma unroll
  for (int ee = 0; ee < NEXP; ++ee) cnt[ee] = chunkOffsets[chunk * NEXP + ee];
  unsigned ltmask = (1u << lane) - 1u;
  int base = chunk * slotsPerChunk;
  for (int i = base + lane; i < base + slotsPerChunk; i += 32) {
    int e = (i < nSlots) ? tokExpert[i] : -1;
    int pos = -1;
#pragma unroll
    for (int ee = 0; ee < NEXP; ++ee) {
      unsigned m = (unsigned)__ballot(e == ee);
      if (e == ee) pos = cnt[ee] + __popc(m & ltmask);
      cnt[ee] += __popc(m);
    }
    if (i < nSlots)
      slotRow[i] = (pos >= 0 && pos < cap) ? (e * capR + pos) : -1;
  }
}

__global__ void wnorm_kernel(const int* __restrict__ slotRow,
                             const float* __restrict__ tokW,
                             float* __restrict__ wNorm, int N) {
  int t = blockIdx.x * blockDim.x + threadIdx.x;
  if (t >= N) return;
  int i0 = 2 * t, i1 = 2 * t + 1;
  float w0 = (slotRow[i0] >= 0) ? tokW[i0] : 0.f;
  float w1 = (slotRow[i1] >= 0) ? tokW[i1] : 0.f;
  float inv = 1.f / (w0 + w1 + 1e-9f);
  wNorm[i0] = w0 * inv;
  wNorm[i1] = w1 * inv;
}

// ---------------- Gather kept tokens into bf16 expert buffers ----------------
__global__ void gather_kernel(const float* __restrict__ x,
                              const int* __restrict__ slotRow,
                              unsigned short* __restrict__ expertIn) {
  int i = blockIdx.x;
  int s = slotRow[i];
  if (s < 0) return;
  int tok = i >> 1;
  int j0 = threadIdx.x * 4;
  float4 v = *(const float4*)(x + (size_t)tok * DMODEL + j0);
  ushort4 o = make_ushort4(f2bf(v.x), f2bf(v.y), f2bf(v.z), f2bf(v.w));
  *(ushort4*)(expertIn + (size_t)s * DMODEL + j0) = o;
}

// ---------------- Weight transpose + f32->bf16 ----------------
// W: [E][Krows][Ncols] f32  ->  Wt: [E][Ncols][Krows] bf16 (K contiguous per column)
__global__ void wtrans_kernel(const float* __restrict__ W,
                              unsigned short* __restrict__ Wt,
                              int Krows, int Ncols) {
  size_t per = (size_t)Krows * Ncols;
  size_t idx = (size_t)blockIdx.x * blockDim.x + threadIdx.x;
  if (idx >= per * NEXP) return;
  size_t e = idx / per, r = idx % per;
  size_t kk = r / Ncols, n = r % Ncols;
  Wt[e * per + n * (size_t)Krows + kk] = f2bf(W[idx]);
}

// ---------------- WMMA bf16 GEMM, TDM double-buffered through LDS ----------------
// Block (8 waves) computes a 128(M) x 64(N) tile. Per K-block of 64:
//   wave0 issues TDM loads for the NEXT tile, then s_wait_tensorcnt 2 so the
//   CURRENT tile's (in-order) loads are complete while the next stays in flight.
// A:  [Mtot, Kdim] bf16 row-major; Bt: [E][Ncols][Kdim] bf16.
__global__ __launch_bounds__(256) void ffn_gemm_kernel(
    const unsigned short* __restrict__ A,
    const unsigned short* __restrict__ Bt,
    const float* __restrict__ bias,
    unsigned short* __restrict__ outBf,
    float* __restrict__ outF,
    int Kdim, int Ncols, int capR, int nBlocksN, int doGelu) {
  __shared__ char smem[2 * BUFSZ];

  int tid    = threadIdx.x;
  int waveId = tid >> 5;
  int lane   = tid & 31;
  int halfId = lane >> 4;
  int l15    = lane & 15;

  int bn = blockIdx.x % nBlocksN;
  int bm = blockIdx.x / nBlocksN;
  int M0 = bm * 128;
  int N0 = bn * 64;
  int e  = M0 / capR;

  const unsigned short* Ab = A + (size_t)M0 * Kdim;
  const unsigned short* Bb = Bt + ((size_t)e * Ncols + N0) * Kdim;

  unsigned ldsBase = (unsigned)(size_t)&smem[0];   // low 32 bits = LDS byte offset

  int waveM = waveId & 3;    // 4 M-strips of 32 rows
  int waveN = waveId >> 2;   // 2 N-strips of 32 cols

  v8f acc[2][2] = {};
  union FragU { v16bf v; uint4 u[2]; };

  int nk = Kdim / KBLK;

  // Prologue: stage tile 0 into buffer 0
  if (waveId == 0) {
    tdm_load_2d(ldsBase, Ab, (unsigned)Kdim, 128u, KBLK, 128u);
    tdm_load_2d(ldsBase + 128u * LDSROW, Bb, (unsigned)Kdim, 64u, KBLK, 64u);
  }

  for (int kb = 0; kb < nk; ++kb) {
    int cur = kb & 1;
    if (waveId == 0) {
      if (kb + 1 < nk) {
        // Issue next tile into the other buffer (its previous consumers finished
        // at the end-of-iteration barrier of kb-1), then wait for current tile:
        // TENSORcnt is in-order, 2 outstanding belong to the next tile.
        unsigned dst = ldsBase + (unsigned)(1 - cur) * BUFSZ;
        int kn = (kb + 1) * KBLK;
        tdm_load_2d(dst, Ab + kn, (unsigned)Kdim, 128u, KBLK, 128u);
        tdm_load_2d(dst + 128u * LDSROW, Bb + kn, (unsigned)Kdim, 64u, KBLK, 64u);
        __builtin_amdgcn_s_wait_tensorcnt(2);
      } else {
        __builtin_amdgcn_s_wait_tensorcnt(0);
      }
    }
    __syncthreads();                         // current tile visible to all waves

    const char* aPtr = &smem[cur * BUFSZ];
    const char* bPtr = aPtr + 128 * LDSROW;
#pragma unroll
    for (int kk = 0; kk < KBLK; kk += 32) {
      FragU a[2], b[2];
#pragma unroll
      for (int mi = 0; mi < 2; ++mi) {
        int row = waveM * 32 + mi * 16 + l15;
        const char* p = aPtr + row * LDSROW + (kk + halfId * 8) * 2;
        a[mi].u[0] = *(const uint4*)p;          // K {0..7 | 8..15}
        a[mi].u[1] = *(const uint4*)(p + 32);   // K {16..23 | 24..31}
      }
#pragma unroll
      for (int ni = 0; ni < 2; ++ni) {
        int col = waveN * 32 + ni * 16 + l15;
        const char* p = bPtr + col * LDSROW + (kk + halfId * 8) * 2;
        b[ni].u[0] = *(const uint4*)p;
        b[ni].u[1] = *(const uint4*)(p + 32);
      }
#pragma unroll
      for (int mi = 0; mi < 2; ++mi)
#pragma unroll
        for (int ni = 0; ni < 2; ++ni)
          acc[mi][ni] = __builtin_amdgcn_wmma_f32_16x16x32_bf16(
              false, a[mi].v, false, b[ni].v, (short)0, acc[mi][ni], false, false);
    }
    __syncthreads();                         // this buffer free for tile kb+2
  }

  // Store: C layout -> vgpr r holds row (base + halfId*8 + r), col = ... + l15
#pragma unroll
  for (int ni = 0; ni < 2; ++ni) {
    int col = N0 + waveN * 32 + ni * 16 + l15;
    float b = bias[(size_t)e * Ncols + col];
#pragma unroll
    for (int mi = 0; mi < 2; ++mi) {
      int rowb = M0 + waveM * 32 + mi * 16 + halfId * 8;
#pragma unroll
      for (int r = 0; r < 8; ++r) {
        float v = acc[mi][ni][r] + b;
        size_t idx = (size_t)(rowb + r) * Ncols + col;
        if (doGelu) {
          v = 0.5f * v * (1.f + erff(v * 0.70710678118654752f));
          outBf[idx] = f2bf(v);
        } else {
          outF[idx] = v;
        }
      }
    }
  }
}

// ---------------- Combine: per-token weighted sum of kept expert rows ----------------
__global__ void combine_kernel(const float* __restrict__ expertOut,
                               const int* __restrict__ slotRow,
                               const float* __restrict__ wNorm,
                               float* __restrict__ y) {
  int tok = blockIdx.x;
  int d0 = threadIdx.x * 4;
  float4 acc = make_float4(0.f, 0.f, 0.f, 0.f);
#pragma unroll
  for (int kk = 0; kk < TOPK; ++kk) {
    int i = TOPK * tok + kk;
    int s = slotRow[i];
    if (s >= 0) {
      float w = wNorm[i];
      float4 v = *(const float4*)(expertOut + (size_t)s * DMODEL + d0);
      acc.x += w * v.x; acc.y += w * v.y; acc.z += w * v.z; acc.w += w * v.w;
    }
  }
  *(float4*)(y + (size_t)tok * DMODEL + d0) = acc;
}

extern "C" void kernel_launch(void* const* d_in, const int* in_sizes, int n_in,
                              void* d_out, int out_size, void* d_ws, size_t ws_size,
                              hipStream_t stream) {
  const float* x  = (const float*)d_in[0];
  const float* Wr = (const float*)d_in[1];
  const float* br = (const float*)d_in[2];
  const float* W1 = (const float*)d_in[3];
  const float* b1 = (const float*)d_in[4];
  const float* W2 = (const float*)d_in[5];
  const float* b2 = (const float*)d_in[6];
  float* y = (float*)d_out;

  int N = in_sizes[0] / DMODEL;                 // 32768 tokens
  int nSlots = N * TOPK;                        // 65536
  int cap  = (int)((5LL * N * TOPK + (4 * NEXP - 1)) / (4 * NEXP));  // ceil(1.25*N*K/E)
  int capR = (cap + 127) & ~127;                // rows padded to 128 for M-tiling
  size_t Mtot = (size_t)NEXP * capR;

  // Carve workspace
  char* ws = (char*)d_ws;
  auto carve = [&](size_t bytes) -> void* {
    void* p = (void*)ws;
    ws += (bytes + 255) & ~(size_t)255;
    return p;
  };
  unsigned short* W1t      = (unsigned short*)carve((size_t)NEXP * DMODEL * DFFN * 2);
  unsigned short* W2t      = (unsigned short*)carve((size_t)NEXP * DFFN * DMODEL * 2);
  unsigned short* expertIn = (unsigned short*)carve(Mtot * DMODEL * 2);
  unsigned short* hBuf     = (unsigned short*)carve(Mtot * DFFN * 2);
  float*          expertOut= (float*)carve(Mtot * DMODEL * 4);
  int*   tokExpert   = (int*)carve((size_t)nSlots * 4);
  float* tokW        = (float*)carve((size_t)nSlots * 4);
  int*   slotRow     = (int*)carve((size_t)nSlots * 4);
  float* wNorm       = (float*)carve((size_t)nSlots * 4);
  int nChunks = 64;
  int slotsPerChunk = ((nSlots + nChunks - 1) / nChunks + 31) & ~31;
  int* chunkCounts  = (int*)carve((size_t)nChunks * NEXP * 4);
  int* chunkOffsets = (int*)carve((size_t)nChunks * NEXP * 4);

  // 0) bf16 weight transposes
  {
    size_t tot = (size_t)NEXP * DMODEL * DFFN;
    int blocks = (int)((tot + 255) / 256);
    wtrans_kernel<<<blocks, 256, 0, stream>>>(W1, W1t, DMODEL, DFFN);
    wtrans_kernel<<<blocks, 256, 0, stream>>>(W2, W2t, DFFN, DMODEL);
  }
  // 1) router
  router_kernel<<<(N + 7) / 8, 256, 0, stream>>>(x, Wr, br, tokExpert, tokW, N);
  // 2) order-exact capacity dispatch
  count_kernel<<<nChunks, 32, 0, stream>>>(tokExpert, chunkCounts, slotsPerChunk, nSlots);
  scan_kernel<<<1, 32, 0, stream>>>(chunkCounts, chunkOffsets, nChunks);
  assign_kernel<<<nChunks, 32, 0, stream>>>(tokExpert, chunkOffsets, slotRow,
                                            slotsPerChunk, nSlots, cap, capR);
  wnorm_kernel<<<(N + 255) / 256, 256, 0, stream>>>(slotRow, tokW, wNorm, N);
  // 3) gather
  gather_kernel<<<nSlots, DMODEL / 4, 0, stream>>>(x, slotRow, expertIn);
  // 4) expert FFN via TDM double-buffered WMMA bf16
  {
    int mBlocks = (int)(Mtot / 128);
    int nB1 = DFFN / 64;
    ffn_gemm_kernel<<<mBlocks * nB1, 256, 0, stream>>>(
        expertIn, W1t, b1, hBuf, nullptr, DMODEL, DFFN, capR, nB1, 1);
    int nB2 = DMODEL / 64;
    ffn_gemm_kernel<<<mBlocks * nB2, 256, 0, stream>>>(
        hBuf, W2t, b2, nullptr, expertOut, DFFN, DMODEL, capR, nB2, 0);
  }
  // 5) combine
  combine_kernel<<<N, DMODEL / 4, 0, stream>>>(expertOut, slotRow, wNorm, y);
}
